// DlrmSmall_52269751992762
// MI455X (gfx1250) — matrix operand order repaired and checked by reading, not compiled, not run on hardware
//
#include <hip/hip_runtime.h>
#include <hip/hip_bf16.h>

typedef _Float16 half_t;
typedef __attribute__((ext_vector_type(16))) _Float16 v16h;
typedef __attribute__((ext_vector_type(8)))  _Float16 v8h;
typedef __attribute__((ext_vector_type(8)))  float    v8f;
typedef __attribute__((ext_vector_type(4)))  unsigned int v4u;
typedef __attribute__((ext_vector_type(8)))  int v8i;
typedef __attribute__((ext_vector_type(4)))  int v4i;

#define BATCH_N   65536
#define VOCAB_N   1048576
#define EMBED_N   128
#define NSPARSE   26
#define TOPIN_LD  512
#define SPB       4   // samples per interaction block

// Assemble a 16-element f16 WMMA fragment from two contiguous 8-element (16B) LDS chunks.
__device__ static inline v16h make_frag(const half_t* lo, const half_t* hi) {
  v8h a = *(const v8h*)lo;
  v8h b = *(const v8h*)hi;
  v16h r;
#pragma unroll
  for (int i = 0; i < 8; ++i) { r[i] = a[i]; r[i + 8] = b[i]; }
  return r;
}

// ---------------------------------------------------------------------------
// TDM: async DMA of a 2-D tile of 16-bit elements, global -> LDS (row-major,
// packed). Descriptor built per CDNA5 ISA:
//   group0: count=1 | lds_addr | global_addr[56:0] | type=2
//   group1: data_size=1(2B), tensor dims == tile dims (no OOB), 48b dim0 stride
//   group2/3 (+ trailing group, clang-23 6-arg form): zero (2-D tensor)
// Tracked by TENSORcnt; caller pairs with __builtin_amdgcn_s_wait_tensorcnt.
// ---------------------------------------------------------------------------
__device__ static inline void tdm_load_2d(unsigned lds_off, const half_t* gptr,
                                          unsigned tile_d0, unsigned tile_d1,
                                          unsigned long long stride0) {
  unsigned long long ga = (unsigned long long)(size_t)gptr;
  v4u g0;
  g0[0] = 1u;                                            // count=1, no gather
  g0[1] = lds_off;                                       // lds_addr (bytes)
  g0[2] = (unsigned)(ga & 0xFFFFFFFFu);                  // global_addr[31:0]
  g0[3] = (unsigned)((ga >> 32) & 0x1FFFFFFu) | (2u << 30); // addr[56:32] | type=2
  v8i g1;
  g1[0] = (int)(1u << 16);                               // data_size=1 (2B), mask=0
  g1[1] = (int)((tile_d0 & 0xFFFFu) << 16);              // tensor_dim0[15:0]
  g1[2] = (int)(((tile_d0 >> 16) & 0xFFFFu) |            // tensor_dim0[31:16]
                ((tile_d1 & 0xFFFFu) << 16));            // tensor_dim1[15:0]
  g1[3] = (int)(((tile_d1 >> 16) & 0xFFFFu) |            // tensor_dim1[31:16]
                ((tile_d0 & 0xFFFFu) << 16));            // tile_dim0
  g1[4] = (int)(tile_d1 & 0xFFFFu);                      // tile_dim1, tile_dim2=0
  g1[5] = (int)(unsigned)(stride0 & 0xFFFFFFFFull);      // dim0_stride[31:0]
  g1[6] = (int)(unsigned)((stride0 >> 32) & 0xFFFFull);  // dim0_stride[47:32]
  g1[7] = 0;                                             // dim1_stride (unused, 2-D)
  v4i z4 = {0, 0, 0, 0};
  v8i z8 = {0, 0, 0, 0, 0, 0, 0, 0};
  __builtin_amdgcn_tensor_load_to_lds(g0, g1, z4, z4, z8, 0);
}

// ---------------------------------------------------------------------------
// Packing kernels: f32 weights (K,N) -> f16 transposed+K-padded (N,Kpad).
// ---------------------------------------------------------------------------
__global__ void pack_weight_kernel(const float* __restrict__ W, half_t* __restrict__ Wp,
                                   int K, int N, int Kpad) {
  int i = blockIdx.x * blockDim.x + threadIdx.x;
  if (i >= N * Kpad) return;
  int n = i / Kpad, k = i % Kpad;
  float v = (k < K) ? W[(size_t)k * N + n] : 0.0f;
  Wp[(size_t)n * Kpad + k] = (half_t)v;
}

__global__ void pack_dense_kernel(const float* __restrict__ D, half_t* __restrict__ A) {
  int i = blockIdx.x * blockDim.x + threadIdx.x;   // over BATCH*32
  int b = i >> 5, k = i & 31;
  A[i] = (half_t)((k < 13) ? D[(size_t)b * 13 + k] : 0.0f);
}

// ---------------------------------------------------------------------------
// Generic f16 WMMA GEMM:  C[m, coff+n] = act( A[m,:K] @ Wp[n,:K]^T + bias[n] )
//   A: f16 row-major, leading dim lda (K <= lda, K % 32 == 0)
//   Wp: f16, pre-transposed [N][K]
//   Block: 256 threads = 8 waves. Tile: M=128 (16 rows/wave), N=64 (4 WMMAs/wave).
//   K tiles are DMA'd by the Tensor Data Mover into double-buffered LDS:
//   wave 0 issues next-tile TDM loads, waits TENSORcnt<=2 so the current
//   tile's two (in-order) DMAs have retired, then the barrier releases compute.
// ---------------------------------------------------------------------------
__global__ __launch_bounds__(256) void gemm_wmma_kernel(
    const half_t* __restrict__ A, int lda,
    const half_t* __restrict__ Wp,
    const float* __restrict__ bias,
    half_t* __restrict__ C, int ldc, int coff,
    int K, int relu)
{
  __shared__ __align__(16) half_t sA[2][128 * 32];
  __shared__ __align__(16) half_t sB[2][64 * 32];

  const int tid  = threadIdx.x;
  const int wave = tid >> 5;          // wave32
  const int lane = tid & 31;
  const size_t m0 = (size_t)blockIdx.x * 128;
  const int n0 = blockIdx.y * 64;

  v8f acc[4] = {};

  // Prologue: wave 0 kicks off async TDM loads of the first K tile.
  if (tid < 32) {
    tdm_load_2d((unsigned)(size_t)&sA[0][0], A + m0 * (size_t)lda, 32, 128,
                (unsigned long long)lda);
    tdm_load_2d((unsigned)(size_t)&sB[0][0], Wp + (size_t)n0 * K, 32, 64,
                (unsigned long long)K);
  }

  int cur = 0;
  for (int k0 = 0; k0 < K; k0 += 32) {
    if (tid < 32) {
      if (k0 + 32 < K) {
        // Issue next tile into the other buffer (all readers of it synced at
        // the end-of-iteration barrier of the previous pass).
        tdm_load_2d((unsigned)(size_t)&sA[cur ^ 1][0],
                    A + m0 * (size_t)lda + k0 + 32, 32, 128,
                    (unsigned long long)lda);
        tdm_load_2d((unsigned)(size_t)&sB[cur ^ 1][0],
                    Wp + (size_t)n0 * K + k0 + 32, 32, 64,
                    (unsigned long long)K);
        __builtin_amdgcn_s_wait_tensorcnt((short)2);  // current tile retired
      } else {
        __builtin_amdgcn_s_wait_tensorcnt((short)0);
      }
    }
    __syncthreads();

    // A fragment: lane<16 holds K {0..7,16..23}, lane>=16 holds K {8..15,24..31}.
    int arow = wave * 16 + (lane & 15);
    int kbA = (lane < 16) ? 0 : 8;
    v16h af = make_frag(&sA[cur][arow * 32 + kbA], &sA[cur][arow * 32 + kbA + 16]);

#pragma unroll
    for (int j = 0; j < 4; ++j) {
      // B fragment: lane<16 holds K {0..7,16..23} of column n, lane>=16 holds K {8..15,24..31}.
      int brow = j * 16 + (lane & 15);
      int kbB = (lane < 16) ? 0 : 8;
      v16h bf = make_frag(&sB[cur][brow * 32 + kbB], &sB[cur][brow * 32 + kbB + 16]);
      acc[j] = __builtin_amdgcn_wmma_f32_16x16x32_f16(
          false, af, false, bf, (short)0, acc[j], false, false);
    }
    __syncthreads();   // all waves done reading before this buffer is re-filled
    cur ^= 1;
  }

  // Epilogue: D layout — vgpr r: lanes 0-15 -> (M=r, N=lane), lanes 16-31 -> (M=r+8, N=lane-16).
  int crow = wave * 16 + ((lane < 16) ? 0 : 8);
  int ccol = lane & 15;
#pragma unroll
  for (int j = 0; j < 4; ++j) {
    float bv = bias[n0 + j * 16 + ccol];
#pragma unroll
    for (int r = 0; r < 8; ++r) {
      float v = acc[j][r] + bv;
      if (relu) v = fmaxf(v, 0.0f);
      C[(m0 + crow + r) * (size_t)ldc + coff + n0 + j * 16 + ccol] = (half_t)v;
    }
  }
}

// ---------------------------------------------------------------------------
// Embedding gather + pairwise interaction (27x27 Gram via WMMA) + top-input build.
// topin[:,0:128] already holds h (f16, written by bottom GEMM 3).
// Writes topin[:,128:506] = triu(Gram), topin[:,506:512] = 0.
// ---------------------------------------------------------------------------
__global__ __launch_bounds__(256) void interact_kernel(
    const int* __restrict__ sidx,
    const float* __restrict__ emb,
    half_t* __restrict__ topin)
{
  __shared__ __align__(16) half_t comb[SPB][32][EMBED_N];  // 32 KB
  __shared__ __align__(16) float  gram[SPB][32][32];       // 16 KB
  const int tid  = threadIdx.x;
  const int wave = tid >> 5;
  const int lane = tid & 31;
  const int base = blockIdx.x * SPB;

  // Gather 26 embedding rows per sample -> combined rows 1..26 (f32 -> f16).
  const int CH_EMB = SPB * NSPARSE * (EMBED_N / 4);
  for (int ch = tid; ch < CH_EMB; ch += 256) {
    int s   = ch / (NSPARSE * 32);
    int rem = ch % (NSPARSE * 32);
    int row = rem >> 5;
    int seg = rem & 31;
    int idx = sidx[(size_t)(base + s) * NSPARSE + row] & (VOCAB_N - 1);
    float4 v = *(const float4*)(emb + (size_t)idx * EMBED_N + seg * 4);
    half_t* dst = &comb[s][row + 1][seg * 4];
    dst[0] = (half_t)v.x; dst[1] = (half_t)v.y;
    dst[2] = (half_t)v.z; dst[3] = (half_t)v.w;
  }
  // Combined row 0 = h (already f16 in topin).
  for (int ch = tid; ch < SPB * (EMBED_N / 8); ch += 256) {
    int s = ch >> 4, seg = ch & 15;
    *(float4*)&comb[s][0][seg * 8] =
        *(const float4*)&topin[(size_t)(base + s) * TOPIN_LD + seg * 8];
  }
  __syncthreads();

  // Gram = combined @ combined^T, only tiles touching the upper triangle:
  // (i,j) in {(0,0),(0,1),(1,1)}; the (1,0) block is strictly below the diagonal.
  for (int job = wave; job < SPB * 3; job += 8) {
    int s = job / 3, t = job % 3;
    int ti = (t == 2) ? 1 : 0;
    int tj = (t == 0) ? 0 : 1;
    int arow = ti * 16 + (lane & 15);
    int brow = tj * 16 + (lane & 15);
    int kbA = (lane < 16) ? 0 : 8;
    int kbB = (lane < 16) ? 0 : 8;
    v8f acc = {};
#pragma unroll
    for (int kk = 0; kk < 4; ++kk) {
      v16h af = make_frag(&comb[s][arow][kk * 32 + kbA],
                          &comb[s][arow][kk * 32 + kbA + 16]);
      v16h bf = make_frag(&comb[s][brow][kk * 32 + kbB],
                          &comb[s][brow][kk * 32 + kbB + 16]);
      acc = __builtin_amdgcn_wmma_f32_16x16x32_f16(
          false, af, false, bf, (short)0, acc, false, false);
    }
    int gr = ti * 16 + ((lane < 16) ? 0 : 8);
    int gc = tj * 16 + (lane & 15);
#pragma unroll
    for (int r = 0; r < 8; ++r) gram[s][gr + r][gc] = acc[r];
  }
  __syncthreads();

  // Upper triangle (incl. diagonal), row-major — matches np.triu_indices(27).
  for (int t = tid; t < SPB * 378; t += 256) {
    int s = t / 378, p = t % 378;
    int r = 0, n = 27, pp = p;
    while (pp >= n) { pp -= n; n--; r++; }
    int c = r + pp;
    topin[(size_t)(base + s) * TOPIN_LD + 128 + p] = (half_t)gram[s][r][c];
  }
  // Zero pad cols 506..511 (K-padding consumed by the first top GEMM).
  if (tid < SPB * 6) {
    int s = tid / 6, c = 506 + tid % 6;
    topin[(size_t)(base + s) * TOPIN_LD + c] = (half_t)0.0f;
  }
}

// ---------------------------------------------------------------------------
// Final 256 -> 1 layer (no ReLU), f32 accumulate, writes d_out.
// ---------------------------------------------------------------------------
__global__ void final_layer_kernel(const half_t* __restrict__ A,
                                   const half_t* __restrict__ w,
                                   const float* __restrict__ bias,
                                   float* __restrict__ out) {
  int b = blockIdx.x * blockDim.x + threadIdx.x;
  const half_t* row = A + (size_t)b * 256;
  float acc = 0.0f;
#pragma unroll 8
  for (int i = 0; i < 256; ++i) acc += (float)row[i] * (float)w[i];
  out[b] = acc + bias[0];
}

// ---------------------------------------------------------------------------
extern "C" void kernel_launch(void* const* d_in, const int* in_sizes, int n_in,
                              void* d_out, int out_size, void* d_ws, size_t ws_size,
                              hipStream_t stream) {
  (void)in_sizes; (void)n_in; (void)out_size; (void)ws_size;
  const float* dense = (const float*)d_in[0];
  const int*   sidx  = (const int*)  d_in[1];
  const float* emb   = (const float*)d_in[2];
  const float* bw0 = (const float*)d_in[3];  const float* bb0 = (const float*)d_in[4];
  const float* bw1 = (const float*)d_in[5];  const float* bb1 = (const float*)d_in[6];
  const float* bw2 = (const float*)d_in[7];  const float* bb2 = (const float*)d_in[8];
  const float* tw0 = (const float*)d_in[9];  const float* tb0 = (const float*)d_in[10];
  const float* tw1 = (const float*)d_in[11]; const float* tb1 = (const float*)d_in[12];
  const float* tw2 = (const float*)d_in[13]; const float* tb2 = (const float*)d_in[14];
  const float* tw3 = (const float*)d_in[15]; const float* tb3 = (const float*)d_in[16];
  const float* tw4 = (const float*)d_in[17]; const float* tb4 = (const float*)d_in[18];
  float* out = (float*)d_out;

  char* ws = (char*)d_ws;
  size_t off = 0;
  auto take = [&](size_t elems) -> half_t* {
    half_t* p = (half_t*)(ws + off);
    off = (off + elems * sizeof(half_t) + 255) & ~(size_t)255;
    return p;
  };
  half_t* wpb0 = take(512 * 32);
  half_t* wpb1 = take(256 * 512);
  half_t* wpb2 = take(128 * 256);
  half_t* wpt0 = take(1024 * 512);
  half_t* wpt1 = take(1024 * 1024);
  half_t* wpt2 = take(512 * 1024);
  half_t* wpt3 = take(256 * 512);
  half_t* wpt4 = take(256);
  half_t* dense16 = take((size_t)BATCH_N * 32);
  half_t* buf0  = take((size_t)BATCH_N * 1024);
  half_t* buf1  = take((size_t)BATCH_N * 1024);
  half_t* topin = take((size_t)BATCH_N * TOPIN_LD);

  auto packW = [&](const float* W, half_t* Wp, int K, int N, int Kpad) {
    int total = N * Kpad;
    pack_weight_kernel<<<(total + 255) / 256, 256, 0, stream>>>(W, Wp, K, N, Kpad);
  };
  packW(bw0, wpb0, 13,   512,  32);
  packW(bw1, wpb1, 512,  256,  512);
  packW(bw2, wpb2, 256,  128,  256);
  packW(tw0, wpt0, 506,  1024, 512);
  packW(tw1, wpt1, 1024, 1024, 1024);
  packW(tw2, wpt2, 1024, 512,  1024);
  packW(tw3, wpt3, 512,  256,  512);
  packW(tw4, wpt4, 256,  1,    256);
  pack_dense_kernel<<<(BATCH_N * 32) / 256, 256, 0, stream>>>(dense, dense16);

  auto gemm = [&](const half_t* A, int lda, const half_t* Wp, const float* bias,
                  half_t* C, int ldc, int coff, int N, int K, int relu) {
    dim3 grid(BATCH_N / 128, N / 64);
    gemm_wmma_kernel<<<grid, 256, 0, stream>>>(A, lda, Wp, bias, C, ldc, coff, K, relu);
  };

  // Bottom MLP: 13(->32) -> 512 -> 256 -> 128 (h written into topin[:,0:128]).
  gemm(dense16, 32,  wpb0, bb0, buf0,  512,  0, 512,  32,   1);
  gemm(buf0,    512, wpb1, bb1, buf1,  256,  0, 256,  512,  1);
  gemm(buf1,    256, wpb2, bb2, topin, 512,  0, 128,  256,  1);

  // Embedding gather + pairwise interaction -> topin[:,128:512].
  interact_kernel<<<BATCH_N / SPB, 256, 0, stream>>>(sidx, emb, topin);

  // Top MLP: 506(->512) -> 1024 -> 1024 -> 512 -> 256 -> 1.
  gemm(topin, 512,  wpt0, tb0, buf0, 1024, 0, 1024, 512,  1);
  gemm(buf0,  1024, wpt1, tb1, buf1, 1024, 0, 1024, 1024, 1);
  gemm(buf1,  1024, wpt2, tb2, buf0, 512,  0, 512,  1024, 1);
  gemm(buf0,  512,  wpt3, tb3, buf1, 256,  0, 256,  512,  1);
  final_layer_kernel<<<BATCH_N / 256, 256, 0, stream>>>(buf1, wpt4, tb4, out);
}
